// MultiScaleFeatures_14224931684667
// MI455X (gfx1250) — matrix operand order repaired and checked by reading, not compiled, not run on hardware
//
#include <hip/hip_runtime.h>
#include <hip/hip_bf16.h>

// ===========================================================================
// SEANet encoder + RVQ for MI455X (gfx1250, wave32, WMMA).
// Conv/GEMM path: 4-wave workgroups, 128x32 output per block. B (activation)
// fragments are gathered cooperatively once per K-chunk into LDS in WMMA
// fragment layout and re-read by all waves as 32B ds loads; A (weights) are
// pre-packed into fragment order (contiguous 32B global loads). 2x2 register
// blocking -> 4 v_wmma_f32_16x16x32_f16 per wave per K-chunk with full A/B
// reuse. Gather modes are template-specialized (no dead branches).
// LSTM: persistent single-WGP kernel, h-state in LDS (f16 A operand),
// fragment-packed Whh streamed from L2, cell state in registers.
// ===========================================================================

typedef __attribute__((ext_vector_type(16))) _Float16 v16h;
typedef __attribute__((ext_vector_type(8)))  _Float16 v8h;
typedef __attribute__((ext_vector_type(8)))  float    v8f;

__device__ __forceinline__ float elu1(float x) {
  return x > 0.f ? x : (__expf(x) - 1.f);
}
__device__ __forceinline__ float sigmf(float x) {
  return 1.f / (1.f + __expf(-x));
}
__device__ __forceinline__ int reflect_idx(int p, int T) {
  if (p < 0) p = -p;
  if (p >= T) p = 2 * T - 2 - p;
  return p;
}

// ---------------------------------------------------------------------------
// Pack a row-major f32 matrix (Rows x Kred) into WMMA fragment order (f16).
// Layout: dst[((tile*kChunks + kc)*32 + lane)*16 + j], zero-filled OOB.
//   astyle=1 : A-matrix lane layout  (K = base8 + (j&7) + (j>>3)*16)
//   astyle=0 : contiguous k order    (K = base16 + j), not used here
// ---------------------------------------------------------------------------
__global__ void pack_frag_kernel(const float* __restrict__ src,
                                 _Float16* __restrict__ dst, int Rows, int Kred,
                                 int rowTiles, int kChunks, int astyle) {
  int i = blockIdx.x * blockDim.x + threadIdx.x;
  if (i >= rowTiles * kChunks * 32) return;
  int lane = i & 31;
  int kc   = (i >> 5) % kChunks;
  int tile = (i >> 5) / kChunks;
  int row  = tile * 16 + (lane & 15);
  _Float16* o = dst + (size_t)i * 16;
#pragma unroll
  for (int j = 0; j < 16; ++j) {
    int r = kc * 32 + (astyle ? (((lane < 16) ? 0 : 8) + (j & 7) + ((j >> 3) << 4))
                              : (((lane < 16) ? 0 : 16) + j));
    float v = (row < Rows && r < Kred) ? src[(size_t)row * Kred + r] : 0.f;
    o[j] = (_Float16)v;
  }
}

// ---------------------------------------------------------------------------
// Fused implicit-GEMM conv / GEMM kernel.
// Block = 128 threads (4 waves) covering 128 output rows x 32 output cols.
// Per K-chunk: 128 threads cooperatively gather the 32x32 B block (fused
// ELU / reflect-pad / im2col) into LDS fragment layout; each active wave then
// does 4 WMMAs (2 M-tiles x 2 N-tiles) from packed-A global + staged-B LDS.
// BMODE: 0 = conv over (B,Cin,Tin), reflect pad, stride (blockIdx.z = b)
//        1 = LSTM x gather: col = t*16+b over (B,C,T)
//        2 = row-major GEMM: src[col*ldb + r]
// ---------------------------------------------------------------------------
template <int BMODE>
__global__ void conv_wmma_kernel(
    const _Float16* __restrict__ Wp, const float* __restrict__ bias,
    const float* __restrict__ src, const float* __restrict__ addsrc,
    float* __restrict__ out,
    int Cout, int Cin, int Kk, int Tin, int Tout,
    int stride, int pl, int ldb, int omode, int ldo, int preact)
{
  __shared__ alignas(32) _Float16 bstage[2][32][16];  // 2 fragments, 2 KB

  const int tid     = threadIdx.x;
  const int lane    = tid & 31;
  const int wave    = tid >> 5;
  const int b       = blockIdx.z;
  const int Kred    = Cin * Kk;
  const int kChunks = (Kred + 31) >> 5;
  const int n       = lane & 15;
  const int colBase = blockIdx.x * 32;

  // cooperative-gather coordinates: 128 threads x 8 halfs = 1024 staged halfs
  const int gf    = tid >> 6;             // which fragment (col group)
  const int gl    = (tid >> 1) & 31;      // fragment lane
  const int gj    = (tid & 1) * 8;        // half-group within lane
  const int gcol  = colBase + gf * 16 + (gl & 15);
  const int gkoff = ((gl < 16) ? 0 : 8) + 2 * gj;

  const int  mTile0     = blockIdx.y * 8 + wave * 2;
  const bool waveActive = (mTile0 * 16) < Cout;  // wave-uniform
  const _Float16* Wp0 = Wp + ((size_t)(mTile0 + 0) * kChunks * 32 + lane) * 16;
  const _Float16* Wp1 = Wp + ((size_t)(mTile0 + 1) * kChunks * 32 + lane) * 16;

  v8f acc00 = {0.f,0.f,0.f,0.f,0.f,0.f,0.f,0.f};
  v8f acc01 = acc00, acc10 = acc00, acc11 = acc00;

  for (int kc = 0; kc < kChunks; ++kc) {
    __syncthreads();  // previous chunk's LDS reads done
    {
      int rbase = kc * 32 + gkoff;
      _Float16* o = &bstage[gf][gl][gj];
      if (BMODE == 2) {
#pragma unroll
        for (int j = 0; j < 8; ++j) {
          int r = rbase + j;
          float v = 0.f;
          if (gcol < Tout && r < Kred) {
            v = src[(size_t)gcol * ldb + r];
            if (preact) v = elu1(v);
          }
          o[j] = (_Float16)v;
        }
      } else if (BMODE == 1) {
        int bb = gcol & 15, tt = gcol >> 4;
#pragma unroll
        for (int j = 0; j < 8; ++j) {
          int r = rbase + j;
          float v = 0.f;
          if (gcol < Tout && r < Kred) {
            v = src[((size_t)bb * Cin + r) * (size_t)Tin + tt];
            if (preact) v = elu1(v);
          }
          o[j] = (_Float16)v;
        }
      } else {
        int ci = rbase / Kk;
        int kk = rbase - ci * Kk;
#pragma unroll
        for (int j = 0; j < 8; ++j) {
          int r = rbase + j;
          float v = 0.f;
          if (gcol < Tout && r < Kred) {
            int p = reflect_idx(gcol * stride + kk - pl, Tin);
            v = src[((size_t)b * Cin + ci) * (size_t)Tin + p];
            if (preact) v = elu1(v);
          }
          o[j] = (_Float16)v;
          if (++kk == Kk) { kk = 0; ++ci; }
        }
      }
    }
    __syncthreads();  // staged B visible to all waves

    if (waveActive) {
      v16h af0 = *(const v16h*)(Wp0 + (size_t)kc * 512);  // 32B contiguous
      v16h af1 = *(const v16h*)(Wp1 + (size_t)kc * 512);
      v16h bf0 = *(const v16h*)&bstage[0][lane][0];       // 32B ds loads
      v16h bf1 = *(const v16h*)&bstage[1][lane][0];
      acc00 = __builtin_amdgcn_wmma_f32_16x16x32_f16(false, af0, false, bf0,
                                                     (short)0, acc00, false, false);
      acc01 = __builtin_amdgcn_wmma_f32_16x16x32_f16(false, af0, false, bf1,
                                                     (short)0, acc01, false, false);
      acc10 = __builtin_amdgcn_wmma_f32_16x16x32_f16(false, af1, false, bf0,
                                                     (short)0, acc10, false, false);
      acc11 = __builtin_amdgcn_wmma_f32_16x16x32_f16(false, af1, false, bf1,
                                                     (short)0, acc11, false, false);
    }
  }

  if (waveActive) {
#pragma unroll
    for (int mi = 0; mi < 2; ++mi) {
#pragma unroll
      for (int ni = 0; ni < 2; ++ni) {
        const v8f& a = mi == 0 ? (ni == 0 ? acc00 : acc01)
                               : (ni == 0 ? acc10 : acc11);
        int col = colBase + ni * 16 + n;
#pragma unroll
        for (int vv = 0; vv < 8; ++vv) {  // D layout: VGPR v -> M=v (+8 hi)
          int mm  = vv + ((lane >> 4) << 3);
          int cco = (mTile0 + mi) * 16 + mm;
          if (cco < Cout && col < Tout) {
            size_t oidx = (omode == 0)
                              ? (((size_t)b * Cout + cco) * (size_t)Tout + col)
                              : ((size_t)col * ldo + cco);
            float val = a[vv];
            if (bias)   val += bias[cco];
            if (addsrc) val += addsrc[oidx];
            out[oidx] = val;
          }
        }
      }
    }
  }
}

// ---------------------------------------------------------------------------
// Persistent recurrent LSTM layer. One workgroup (32 waves), loops T steps.
// g(16x2048) = h(16x512, f16 in LDS) @ Whh^T (fragment-packed, L2-resident).
// Each wave owns 4 N-tiles; A fragment loaded once per K-chunk (2x ds b128)
// and reused across the 4 WMMAs. Cell state in registers, gates in LDS.
// ---------------------------------------------------------------------------
__global__ void lstm_rec_kernel(
    const float* __restrict__ Xp, const _Float16* __restrict__ Whhp,
    const float* __restrict__ bih, const float* __restrict__ bhh,
    float* __restrict__ ys, int T)
{
  extern __shared__ char smem[];
  float*    gbuf = (float*)smem;                                  // 16 x 2048
  _Float16* hbuf = (_Float16*)(smem + 16 * 2048 * sizeof(float)); // 16 x 512

  const int tid   = threadIdx.x;
  const int lane  = tid & 31;
  const int wave  = tid >> 5;
  const int m     = lane & 15;
  const int abase = (lane < 16) ? 0 : 8;

  for (int e = tid; e < 16 * 512; e += 1024) hbuf[e] = (_Float16)0.f;
  float creg[8];
#pragma unroll
  for (int j = 0; j < 8; ++j) creg[j] = 0.f;
  __syncthreads();

  for (int t = 0; t < T; ++t) {
    v8f acc[4];
#pragma unroll
    for (int q = 0; q < 4; ++q) acc[q] = (v8f){0.f,0.f,0.f,0.f,0.f,0.f,0.f,0.f};

    for (int kc = 0; kc < 16; ++kc) {
      // A fragment from LDS: two aligned 16B vector loads
      v8h lo = *(const v8h*)&hbuf[m * 512 + kc * 32 + abase];
      v8h hi = *(const v8h*)&hbuf[m * 512 + kc * 32 + abase + 16];
      v16h af;
#pragma unroll
      for (int j = 0; j < 8; ++j) { af[j] = lo[j]; af[8 + j] = hi[j]; }
#pragma unroll
      for (int q = 0; q < 4; ++q) {
        int nt = wave * 4 + q;
        v16h bf = *(const v16h*)(Whhp + (((size_t)nt * 16 + kc) * 32 + lane) * 16);
        acc[q] = __builtin_amdgcn_wmma_f32_16x16x32_f16(
            false, af, false, bf, (short)0, acc[q], false, false);
      }
    }
#pragma unroll
    for (int q = 0; q < 4; ++q) {
      int nt = wave * 4 + q;
#pragma unroll
      for (int vv = 0; vv < 8; ++vv) {
        int mm = vv + ((lane >> 4) << 3);
        gbuf[mm * 2048 + nt * 16 + (lane & 15)] = acc[q][vv];
      }
    }
    __syncthreads();

    // pointwise state update: 8192 (b,h) states over 1024 threads
#pragma unroll
    for (int j = 0; j < 8; ++j) {
      int e = tid + j * 1024;
      int bb = e >> 9, hh = e & 511;
      size_t row = (size_t)(t * 16 + bb);
      const float* xr = Xp + row * 2048;
      float gi = gbuf[bb * 2048 + hh]        + xr[hh]        + bih[hh]        + bhh[hh];
      float gf = gbuf[bb * 2048 + 512 + hh]  + xr[512 + hh]  + bih[512 + hh]  + bhh[512 + hh];
      float gg = gbuf[bb * 2048 + 1024 + hh] + xr[1024 + hh] + bih[1024 + hh] + bhh[1024 + hh];
      float go = gbuf[bb * 2048 + 1536 + hh] + xr[1536 + hh] + bih[1536 + hh] + bhh[1536 + hh];
      float c = sigmf(gf) * creg[j] + sigmf(gi) * tanhf(gg);
      float h = sigmf(go) * tanhf(c);
      creg[j] = c;
      ys[row * 512 + hh] = h;
      hbuf[bb * 512 + hh] = (_Float16)h;
    }
    __syncthreads();
  }
}

// --------------------------- small helper kernels --------------------------
__global__ void copy_kernel(const float* __restrict__ s, float* __restrict__ d,
                            size_t n) {
  size_t i = (size_t)blockIdx.x * blockDim.x + threadIdx.x;
  if (i < n) d[i] = s[i];
}

// zsum[b,c,t] = emb[b,c,t] + ys2[(t*16+b)*512 + c]   (residual LSTM skip)
__global__ void zsum_kernel(const float* __restrict__ emb,
                            const float* __restrict__ ys2,
                            float* __restrict__ out) {
  size_t i = (size_t)blockIdx.x * blockDim.x + threadIdx.x;
  if (i >= (size_t)16 * 512 * 320) return;
  int b = (int)(i / (512 * 320));
  int rem = (int)(i % (512 * 320));
  int c = rem / 320, t = rem % 320;
  out[i] = emb[i] + ys2[((size_t)(t * 16 + b)) * 512 + c];
}

// cn[bin] = 0.5 * ||codebook row||^2 over all 4*1024 bins
__global__ void cnorm_kernel(const float* __restrict__ cb,
                             float* __restrict__ cn) {
  int bin = blockIdx.x * blockDim.x + threadIdx.x;
  if (bin >= 4096) return;
  const float* r = cb + (size_t)bin * 512;
  float s = 0.f;
  for (int d = 0; d < 512; ++d) s += r[d] * r[d];
  cn[bin] = 0.5f * s;
}

// xd[b,d,ls] = mean over s of residual[b,d,ls*s + k]
__global__ void avgpool_kernel(const float* __restrict__ res,
                               float* __restrict__ xd, int Ls, int s) {
  size_t i = (size_t)blockIdx.x * blockDim.x + threadIdx.x;
  size_t n = (size_t)16 * 512 * Ls;
  if (i >= n) return;
  int b = (int)(i / ((size_t)512 * Ls));
  int rem = (int)(i % ((size_t)512 * Ls));
  int d = rem / Ls, ls = rem % Ls;
  const float* p = res + ((size_t)b * 512 + d) * 320 + (size_t)ls * s;
  float acc = 0.f;
  for (int k = 0; k < s; ++k) acc += p[k];
  xd[i] = acc / (float)s;
}

// argmin over 1024 bins of (0.5||c||^2 - x.c); one thread per (b,ls)
__global__ void argmin_kernel(const float* __restrict__ G,
                              const float* __restrict__ cn, int cb_i, int Ls,
                              int* __restrict__ codes,
                              float* __restrict__ code_out) {
  int i = blockIdx.x * blockDim.x + threadIdx.x;
  if (i >= 16 * Ls) return;
  int b = i / Ls, ls = i % Ls;
  const float* g = G + (size_t)b * 1024 * Ls + ls;
  const float* c = cn + cb_i * 1024;
  float best = c[0] - g[0];
  int bi = 0;
  for (int bin = 1; bin < 1024; ++bin) {
    float sc = c[bin] - g[(size_t)bin * Ls];
    if (sc < best) { best = sc; bi = bin; }
  }
  codes[i] = bi;
  code_out[i] = (float)bi;
}

// quantized += upsample(q); residual -= upsample(q)
__global__ void rvq_update_kernel(const float* __restrict__ cb,
                                  const int* __restrict__ codes, int cb_i,
                                  int s, int Ls, float* __restrict__ quant,
                                  float* __restrict__ resid) {
  size_t i = (size_t)blockIdx.x * blockDim.x + threadIdx.x;
  if (i >= (size_t)16 * 512 * 320) return;
  int b = (int)(i / (512 * 320));
  int rem = (int)(i % (512 * 320));
  int d = rem / 320, l = rem % 320;
  int code = codes[b * Ls + (l / s)];
  float q = cb[((size_t)cb_i * 1024 + code) * 512 + d];
  quant[i] += q;
  resid[i] -= q;
}

// commit += mean over (B*Ls, D) of (xd - q)^2
__global__ void commit_kernel(const float* __restrict__ xd,
                              const float* __restrict__ cb,
                              const int* __restrict__ codes, int cb_i, int Ls,
                              float* __restrict__ commit) {
  int i = blockIdx.x * blockDim.x + threadIdx.x;
  if (i >= 16 * Ls) return;
  int b = i / Ls, ls = i % Ls;
  const float* q = cb + ((size_t)cb_i * 1024 + codes[i]) * 512;
  float acc = 0.f;
  for (int d = 0; d < 512; ++d) {
    float diff = xd[((size_t)b * 512 + d) * Ls + ls] - q[d];
    acc += diff * diff;
  }
  atomicAdd(commit, acc / (16.f * Ls * 512.f));
}

// ===========================================================================
extern "C" void kernel_launch(void* const* d_in, const int* in_sizes, int n_in,
                              void* d_out, int out_size, void* d_ws,
                              size_t ws_size, hipStream_t stream) {
  (void)in_sizes; (void)n_in; (void)ws_size;

  // ---- input pointers (setup_inputs flattening order) ----
  int di = 0;
  const float* audio      = (const float*)d_in[di++];
  const float* conv_in_w  = (const float*)d_in[di++];
  const float* conv_in_b  = (const float*)d_in[di++];
  const float *res_w1[4], *res_b1[4], *res_w2[4], *res_b2[4];
  const float *skip_w[4], *skip_b[4], *down_w[4], *down_b[4];
  for (int s = 0; s < 4; ++s) {
    res_w1[s] = (const float*)d_in[di++]; res_b1[s] = (const float*)d_in[di++];
    res_w2[s] = (const float*)d_in[di++]; res_b2[s] = (const float*)d_in[di++];
    skip_w[s] = (const float*)d_in[di++]; skip_b[s] = (const float*)d_in[di++];
    down_w[s] = (const float*)d_in[di++]; down_b[s] = (const float*)d_in[di++];
  }
  const float *Wih[2], *Whh[2], *bih[2], *bhh[2];
  for (int l = 0; l < 2; ++l) {
    Wih[l] = (const float*)d_in[di++]; Whh[l] = (const float*)d_in[di++];
    bih[l] = (const float*)d_in[di++]; bhh[l] = (const float*)d_in[di++];
  }
  const float* conv_out_w = (const float*)d_in[di++];
  const float* conv_out_b = (const float*)d_in[di++];
  const float* codebooks  = (const float*)d_in[di++];
  // bandwidth_id -> RAND_NUM_QUANT[0] == 4 always

  float* out = (float*)d_out;

  // ---- workspace carve-out ----
  char* ws = (char*)d_ws;
  size_t off = 0;
  auto allocB = [&](size_t bytes) -> char* {
    char* p = ws + off;
    off = (off + bytes + 255) & ~(size_t)255;
    return p;
  };
  float* buf0 = (float*)allocB((size_t)52428800 * 4);  // B*C*T ping
  float* buf1 = (float*)allocB((size_t)52428800 * 4);  // B*C*T pong + tail bufs
  float* buf2 = (float*)allocB((size_t)26214400 * 4);  // half-channel / Xproj
  float* cn      = (float*)allocB(4096 * 4);
  int*   codes_i = (int*)allocB(5120 * 4);

  const size_t EMBN = (size_t)16 * 512 * 320;  // 2,621,440
  float* ys1   = buf1 + (size_t) 0;
  float* ys2   = buf1 + (size_t) 4 * 1024 * 1024;
  float* zsum  = buf1 + (size_t) 8 * 1024 * 1024;
  float* embq  = buf1 + (size_t)12 * 1024 * 1024;
  float* resid = buf1 + (size_t)16 * 1024 * 1024;
  float* xd    = buf1 + (size_t)20 * 1024 * 1024;
  float* G     = buf1 + (size_t)24 * 1024 * 1024;  // 16*1024*320 max

  // ---- pack stationary GEMM operands into WMMA fragment order ----
  auto packA = [&](const float* s, int Rows, int Kred) -> _Float16* {
    int rt = 8 * ((Rows + 127) / 128), kc = (Kred + 31) / 32;  // pad to block M
    size_t frag = (size_t)rt * kc * 32 * 16;
    _Float16* d = (_Float16*)allocB(frag * 2);
    int threads = rt * kc * 32;
    pack_frag_kernel<<<dim3((threads + 255) / 256), dim3(256), 0, stream>>>(
        s, d, Rows, Kred, rt, kc, 1);
    return d;
  };
  auto packB = [&](const float* s, int Rows, int Kred) -> _Float16* {
    int rt = (Rows + 15) / 16, kc = (Kred + 31) / 32;
    size_t frag = (size_t)rt * kc * 32 * 16;
    _Float16* d = (_Float16*)allocB(frag * 2);
    int threads = rt * kc * 32;
    pack_frag_kernel<<<dim3((threads + 255) / 256), dim3(256), 0, stream>>>(
        s, d, Rows, Kred, rt, kc, 1);
    return d;
  };

  _Float16* conv_in_p = packA(conv_in_w, 32, 7);
  _Float16 *rw1p[4], *rw2p[4], *skwp[4], *dwp[4];
  const int Cs[4]   = {32, 64, 128, 256};
  const int Rs[4]   = {2, 4, 5, 8};
  const int TinS[4] = {102400, 51200, 12800, 2560};
  for (int s = 0; s < 4; ++s) {
    int C = Cs[s], R = Rs[s];
    rw1p[s] = packA(res_w1[s], C / 2, C * 3);
    rw2p[s] = packA(res_w2[s], C, C / 2);
    skwp[s] = packA(skip_w[s], C, C);
    dwp[s]  = packA(down_w[s], 2 * C, C * 2 * R);
  }
  _Float16 *Wihp[2], *Whhp[2];
  for (int l = 0; l < 2; ++l) {
    Wihp[l] = packA(Wih[l], 2048, 512);
    Whhp[l] = packB(Whh[l], 2048, 512);   // B operand of recurrent GEMM
  }
  _Float16* conv_out_p = packA(conv_out_w, 512, 512 * 7);
  _Float16* cbp[4];
  for (int i = 0; i < 4; ++i)
    cbp[i] = packA(codebooks + (size_t)i * 1024 * 512, 1024, 512);

  cnorm_kernel<<<dim3(16), dim3(256), 0, stream>>>(codebooks, cn);

  // ---- WMMA conv/GEMM launcher: 4 waves/block, 128x32 output per block ----
  auto conv = [&](const _Float16* W, const float* bias, const float* src,
                  const float* add, float* o, int Cout, int Cin, int Kk,
                  int Tin, int Tout, int stride, int pl, int bmode, int ldb,
                  int omode, int ldo, int preact, int batches) {
    dim3 grid((Tout + 31) / 32, (Cout + 127) / 128, batches);
    if (bmode == 0)
      conv_wmma_kernel<0><<<grid, dim3(128), 0, stream>>>(
          W, bias, src, add, o, Cout, Cin, Kk, Tin, Tout, stride, pl, ldb,
          omode, ldo, preact);
    else if (bmode == 1)
      conv_wmma_kernel<1><<<grid, dim3(128), 0, stream>>>(
          W, bias, src, add, o, Cout, Cin, Kk, Tin, Tout, stride, pl, ldb,
          omode, ldo, preact);
    else
      conv_wmma_kernel<2><<<grid, dim3(128), 0, stream>>>(
          W, bias, src, add, o, Cout, Cin, Kk, Tin, Tout, stride, pl, ldb,
          omode, ldo, preact);
  };

  // ---- encoder front: conv7 (1 -> 32), reflect pad 3 ----
  conv(conv_in_p, conv_in_b, audio, nullptr, buf0,
       32, 1, 7, 102400, 102400, 1, 3, 0, 0, 0, 0, 0, 16);

  // ---- 4 stages: residual unit (true_skip=False) + strided downsample ----
  for (int s = 0; s < 4; ++s) {
    int C = Cs[s], R = Rs[s], T = TinS[s], To = T / R;
    // y1 = conv_k3(elu(h)) : C -> C/2
    conv(rw1p[s], res_b1[s], buf0, nullptr, buf2,
         C / 2, C, 3, T, T, 1, 1, 0, 0, 0, 0, 1, 16);
    // y2 = conv_k1(elu(y1)) : C/2 -> C   (into buf1)
    conv(rw2p[s], res_b2[s], buf2, nullptr, buf1,
         C, C / 2, 1, T, T, 1, 0, 0, 0, 0, 0, 1, 16);
    // h' = skip_conv_k1(h) + y2         (read-modify-write same buffer)
    conv(skwp[s], skip_b[s], buf0, buf1, buf1,
         C, C, 1, T, T, 1, 0, 0, 0, 0, 0, 0, 16);
    // h = conv_k2R_strideR(elu(h')) : C -> 2C
    conv(dwp[s], down_b[s], buf1, nullptr, buf0,
         2 * C, C, 2 * R, T, To, R, R / 2, 0, 0, 0, 0, 1, 16);
  }
  // buf0 now holds pre-LSTM embedding (16, 512, 320)

  // ---- 2-layer residual LSTM ----
  const size_t lstm_lds = 16 * 2048 * sizeof(float) + 16 * 512 * sizeof(_Float16);
  // layer 0: input projection (gather x_t from (B,C,T)) then recurrence
  conv(Wihp[0], nullptr, buf0, nullptr, buf2,
       2048, 512, 1, 320, 5120, 1, 0, 1, 0, 1, 2048, 0, 1);
  lstm_rec_kernel<<<dim3(1), dim3(1024), lstm_lds, stream>>>(
      buf2, Whhp[0], bih[0], bhh[0], ys1, 320);
  // layer 1: row-major input projection from ys1, then recurrence
  conv(Wihp[1], nullptr, ys1, nullptr, buf2,
       2048, 512, 1, 0, 5120, 1, 0, 2, 512, 1, 2048, 0, 1);
  lstm_rec_kernel<<<dim3(1), dim3(1024), lstm_lds, stream>>>(
      buf2, Whhp[1], bih[1], bhh[1], ys2, 320);

  // z = emb + y2 ; emb = conv7(elu(z)) : 512 -> 512
  zsum_kernel<<<dim3((unsigned)((EMBN + 255) / 256)), dim3(256), 0, stream>>>(
      buf0, ys2, zsum);
  conv(conv_out_p, conv_out_b, zsum, nullptr, embq,
       512, 512, 7, 320, 320, 1, 3, 0, 0, 0, 0, 1, 16);

  // ---- multi-scale residual VQ ----
  hipMemsetAsync(d_out, 0, (size_t)out_size * sizeof(float), stream);
  copy_kernel<<<dim3((unsigned)((EMBN + 255) / 256)), dim3(256), 0, stream>>>(
      embq, resid, EMBN);

  const int scales[4] = {8, 4, 2, 1};
  size_t code_off = EMBN;
  float* commit_ptr = out + EMBN + 9600;  // quantized + all codes, then commit
  for (int i = 0; i < 4; ++i) {
    int s = scales[i], Ls = 320 / s;
    size_t nxd = (size_t)16 * 512 * Ls;
    avgpool_kernel<<<dim3((unsigned)((nxd + 255) / 256)), dim3(256), 0,
                     stream>>>(resid, xd, Ls, s);
    // dot(x, codebook): (1024 x 512) @ (512 x Ls) per batch via WMMA
    conv(cbp[i], nullptr, xd, nullptr, G,
         1024, 512, 1, Ls, Ls, 1, 0, 0, 0, 0, 0, 0, 16);
    argmin_kernel<<<dim3((16 * Ls + 127) / 128), dim3(128), 0, stream>>>(
        G, cn, i, Ls, codes_i, out + code_off);
    rvq_update_kernel<<<dim3((unsigned)((EMBN + 255) / 256)), dim3(256), 0,
                        stream>>>(codebooks, codes_i, i, s, Ls, out, resid);
    commit_kernel<<<dim3((16 * Ls + 127) / 128), dim3(128), 0, stream>>>(
        xd, codebooks, codes_i, i, Ls, commit_ptr);
    code_off += (size_t)16 * Ls;
  }
}
